// LoSoU_27676769255707
// MI455X (gfx1250) — matrix operand order repaired and checked by reading, compiled, not run on hardware
//
#include <hip/hip_runtime.h>
#include <hip/hip_bf16.h>
#include <math.h>

// ---------------------------------------------------------------------------
// Types for gfx1250 WMMA (wave32)
// ---------------------------------------------------------------------------
typedef __bf16 bf16_t;
typedef __attribute__((ext_vector_type(8)))  __bf16 bfx8;
typedef __attribute__((ext_vector_type(16))) __bf16 v16bf;
typedef __attribute__((ext_vector_type(8)))  float  v8f;

// dims (compile-time; reference fixes B=4,S=4096,D=2048,F=128)
#define MTOT 16384
#define DDIM 2048
#define FDIM 128
#define HDIM 1024
#define SDIM 4096
#define BDIM 4

enum { EP_SIGMOID = 0, EP_SIGMOID_MUL = 1, EP_GELU_BF16 = 2, EP_BIAS = 3, EP_BIAS_RESID = 4 };

__device__ __forceinline__ float sigmoidf_(float x) { return 1.0f / (1.0f + expf(-x)); }

__device__ __forceinline__ float geluf_(float x) {
  // jax.nn.gelu default (approximate=True, tanh form)
  float x3 = x * x * x;
  return 0.5f * x * (1.0f + tanhf(0.7978845608028654f * (x + 0.044715f * x3)));
}

__device__ __forceinline__ unsigned short f2bf(float f) {
  unsigned u = __builtin_bit_cast(unsigned, f);
  u += 0x7fffu + ((u >> 16) & 1u);  // round-to-nearest-even
  return (unsigned short)(u >> 16);
}

// LDS byte-offset of a generic pointer into a __shared__ object
__device__ __forceinline__ unsigned lds_off(const void* p) {
  return (unsigned)(unsigned long long)(const __attribute__((address_space(3))) void*)p;
}

// Async DMA global -> LDS, 16B per lane, tracked by ASYNCcnt (CDNA5).
__device__ __forceinline__ void async_load_b128(unsigned lds, const void* gptr) {
  unsigned long long g = (unsigned long long)gptr;
  asm volatile("global_load_async_to_lds_b128 %0, %1, off" ::"v"(lds), "v"(g) : "memory");
}
__device__ __forceinline__ void wait_async0() {
  asm volatile("s_wait_asynccnt 0x0" ::: "memory");
}

// ---------------------------------------------------------------------------
// f32 -> bf16 conversion (x / activations path)
// ---------------------------------------------------------------------------
__global__ void f32_to_bf16_kernel(const float* __restrict__ src,
                                   unsigned short* __restrict__ dst, size_t n) {
  size_t id = (size_t)blockIdx.x * blockDim.x + threadIdx.x;
  if (id < n) dst[id] = f2bf(src[id]);
}

// ---------------------------------------------------------------------------
// f32 [K][N] -> bf16 transposed [N][K] (weights, one-time). 32x32 LDS tiles.
// ---------------------------------------------------------------------------
__global__ __launch_bounds__(256) void f32_to_bf16_T_kernel(
    const float* __restrict__ src, unsigned short* __restrict__ dst, int K, int N) {
  __shared__ float tile[32][33];
  int n0 = blockIdx.x * 32, k0 = blockIdx.y * 32;
  int tx = threadIdx.x & 31, ty = threadIdx.x >> 5;  // 32 x 8
#pragma unroll
  for (int i = 0; i < 32; i += 8)
    tile[ty + i][tx] = src[(size_t)(k0 + ty + i) * N + n0 + tx];
  __syncthreads();
#pragma unroll
  for (int i = 0; i < 32; i += 8)
    dst[(size_t)(n0 + ty + i) * K + k0 + tx] = f2bf(tile[tx][ty + i]);
}

// ---------------------------------------------------------------------------
// bf16 WMMA GEMM:  C[M,N] = epilogue(A[M,K] @ W + bias), W given TRANSPOSED
// as Wt[N][K]. Workgroup tile 64x128, wave tile 32x32 (4 x
// v_wmma_f32_16x16x32_bf16 per K-step), BK=32.
//
// Both A and B tiles are row-major contiguous copies -> staged straight into
// LDS with global_load_async_to_lds_b128 (ASYNCcnt), double buffered:
//   [wait asynccnt 0][barrier][issue tile t+1 -> buf^1][compute tile t]
// Fragments are two contiguous ds_load_b128 per operand matching the ISA
// 16-bit A(16x32)/B(32x16) VGPR layouts.
// ---------------------------------------------------------------------------
template <int EP>
__global__ __launch_bounds__(256) void gemm_bf16_wmma(
    const bf16_t* __restrict__ A, const bf16_t* __restrict__ Wt,
    const float* __restrict__ bias, void* __restrict__ Cv,
    const float* __restrict__ aux, int N, int K) {
  constexpr int BM = 64, BN = 128, BK = 32, LA = 40, LB = 40;  // padded strides
  __shared__ __attribute__((aligned(16))) bf16_t As[2][BM * LA];  // 2 x  5 KB
  __shared__ __attribute__((aligned(16))) bf16_t Bs[2][BN * LB];  // 2 x 10 KB

  const int tid  = threadIdx.x;
  const int lane = tid & 31;
  const int wid  = tid >> 5;         // 8 waves
  const int mw   = (wid & 1) * 32;   // wave M offset in block tile
  const int nw   = (wid >> 1) * 32;  // wave N offset in block tile
  const size_t blockM = (size_t)blockIdx.y * BM;
  const int    blockN = blockIdx.x * BN;

  // A fragment: lane%16 = M row; lanes 0-15 hold K 0-7 & 16-23,
  // lanes 16-31 hold K 8-15 & 24-31  -> two contiguous b128 LDS loads.
  const int mA    = lane & 15;
  const int kAoff = (lane >> 4) * 8;
  // B fragment: lane%16 = N col; same K split as A (K 0-7 & 16-23 / 8-15 & 24-31).
  const int nB    = lane & 15;
  const int kBoff = (lane >> 4) * 8;

  // staging split: A: 64 rows x 4 groups of 8 k; B: 128 rows x 4 groups of 8 k
  const int arow = tid >> 2, akg = tid & 3;

  v8f acc[2][2] = {};

  auto issue_tile = [&](int buf, int k0) {
    async_load_b128(lds_off(&As[buf][arow * LA + akg * 8]),
                    A + (blockM + arow) * (size_t)K + k0 + akg * 8);
#pragma unroll
    for (int it = 0; it < 2; ++it) {
      int idx = tid + it * 256;      // 0..511
      int bn  = idx >> 2;            // 0..127
      int bkg = idx & 3;             // 0..3
      async_load_b128(lds_off(&Bs[buf][bn * LB + bkg * 8]),
                      Wt + (size_t)(blockN + bn) * K + k0 + bkg * 8);
    }
  };

  const int nT = K / BK;
  issue_tile(0, 0);

  for (int t = 0; t < nT; ++t) {
    wait_async0();      // this wave's async stores to LDS have landed
    __syncthreads();    // everyone's have; previous buffer free for reuse
    if (t + 1 < nT) issue_tile((t + 1) & 1, (t + 1) * BK);

    const bf16_t* as = As[t & 1];
    const bf16_t* bs = Bs[t & 1];
    union Fr { v16bf v; bfx8 h[2]; };
    Fr a[2], b[2];
#pragma unroll
    for (int i = 0; i < 2; ++i) {
      const bf16_t* pa = &as[(mw + i * 16 + mA) * LA + kAoff];
      a[i].h[0] = *(const bfx8*)pa;         // K 0-7   (or 8-15)
      a[i].h[1] = *(const bfx8*)(pa + 16);  // K 16-23 (or 24-31)
    }
#pragma unroll
    for (int j = 0; j < 2; ++j) {
      const bf16_t* pb = &bs[(nw + j * 16 + nB) * LB + kBoff];
      b[j].h[0] = *(const bfx8*)pb;
      b[j].h[1] = *(const bfx8*)(pb + 16);
    }
#pragma unroll
    for (int i = 0; i < 2; ++i)
#pragma unroll
      for (int j = 0; j < 2; ++j)
        acc[i][j] = __builtin_amdgcn_wmma_f32_16x16x32_bf16(
            false, a[i].v, false, b[j].v, (short)0, acc[i][j], false, false);
  }

  // epilogue: C/D frag layout: lane%16 = col, (lane/16)*8 + v = row
  const int rb = (lane >> 4) * 8;
  const int cc = lane & 15;
#pragma unroll
  for (int i = 0; i < 2; ++i) {
#pragma unroll
    for (int j = 0; j < 2; ++j) {
#pragma unroll
      for (int v = 0; v < 8; ++v) {
        size_t r = blockM + mw + i * 16 + rb + v;
        int    c = blockN + nw + j * 16 + cc;
        float val = acc[i][j][v] + bias[c];
        size_t o = r * (size_t)N + c;
        if constexpr (EP == EP_SIGMOID) {
          ((float*)Cv)[o] = sigmoidf_(val);
        } else if constexpr (EP == EP_SIGMOID_MUL) {
          ((float*)Cv)[o] = sigmoidf_(val) * aux[o];
        } else if constexpr (EP == EP_GELU_BF16) {
          ((unsigned short*)Cv)[o] = f2bf(geluf_(val));
        } else if constexpr (EP == EP_BIAS) {
          ((float*)Cv)[o] = val;
        } else {  // EP_BIAS_RESID
          ((float*)Cv)[o] = val + aux[o];
        }
      }
    }
  }
}

// ---------------------------------------------------------------------------
// EMA over time, in place: 512 independent (b,f) chains. ema[0] = score[0].
// ---------------------------------------------------------------------------
__global__ void ema_kernel(float* __restrict__ score) {
  int id = blockIdx.x * blockDim.x + threadIdx.x;
  if (id >= BDIM * FDIM) return;
  int b = id / FDIM, f = id % FDIM;
  float* p = score + (size_t)b * SDIM * FDIM + f;
  float prev = p[0];
  for (int t = 1; t < SDIM; ++t) {
    prev = 0.15f * p[(size_t)t * FDIM] + 0.85f * prev;
    p[(size_t)t * FDIM] = prev;
  }
}

// ---------------------------------------------------------------------------
// per-row (B*S rows of 128): mean -> denom -> clip -> * V -> bf16 x_comb
// one wave per row (lane covers f, f+32, f+64, f+96)
// ---------------------------------------------------------------------------
__global__ __launch_bounds__(256) void clipmul_kernel(
    const float* __restrict__ ema, const float* __restrict__ V,
    unsigned short* __restrict__ xcomb) {
  int wid = threadIdx.x >> 5, lane = threadIdx.x & 31;
  size_t row = (size_t)blockIdx.x * 8 + wid;
  const float* er = ema + row * FDIM;
  float e[4], sum = 0.0f;
#pragma unroll
  for (int i = 0; i < 4; ++i) { e[i] = er[lane + 32 * i]; sum += e[i]; }
#pragma unroll
  for (int o = 16; o; o >>= 1) sum += __shfl_xor(sum, o);
  float denom = fmaxf(sum * (1.0f / 128.0f), 1e-6f);
#pragma unroll
  for (int i = 0; i < 4; ++i) {
    float sc = fminf(fmaxf(e[i] / denom, -5.0f), 5.0f);
    xcomb[row * FDIM + lane + 32 * i] = f2bf(sc * V[row * FDIM + lane + 32 * i]);
  }
}

// ---------------------------------------------------------------------------
// SiLU gate: gated[r,c] = silu(out[r,c]) * out[r,c+1024]  -> bf16
// ---------------------------------------------------------------------------
__global__ void gate_kernel(const float* __restrict__ out,
                            unsigned short* __restrict__ gated) {
  size_t id = (size_t)blockIdx.x * blockDim.x + threadIdx.x;
  if (id >= (size_t)MTOT * HDIM) return;
  size_t row = id >> 10;
  int    c   = (int)(id & 1023);
  float a  = out[row * DDIM + c];
  float bb = out[row * DDIM + HDIM + c];
  gated[id] = f2bf((a / (1.0f + expf(-a))) * bb);
}

// ---------------------------------------------------------------------------
// LayerNorm over D=2048 per row (h already includes residual)
// ---------------------------------------------------------------------------
__global__ __launch_bounds__(256) void ln_kernel(
    const float* __restrict__ h, const float* __restrict__ gamma,
    const float* __restrict__ beta, float* __restrict__ out) {
  __shared__ float r1[8], r2[8];
  size_t row = blockIdx.x;
  int tid = threadIdx.x, lane = tid & 31, wid = tid >> 5;
  const float* hr = h + row * DDIM;
  float s = 0.0f, s2 = 0.0f;
#pragma unroll
  for (int i = 0; i < 8; ++i) {
    float v = hr[tid + i * 256];
    s += v; s2 += v * v;
  }
#pragma unroll
  for (int o = 16; o; o >>= 1) { s += __shfl_xor(s, o); s2 += __shfl_xor(s2, o); }
  if (lane == 0) { r1[wid] = s; r2[wid] = s2; }
  __syncthreads();
  s = 0.0f; s2 = 0.0f;
#pragma unroll
  for (int w = 0; w < 8; ++w) { s += r1[w]; s2 += r2[w]; }
  float mu  = s * (1.0f / DDIM);
  float var = s2 * (1.0f / DDIM) - mu * mu;
  float inv = rsqrtf(var + 1e-5f);
#pragma unroll
  for (int i = 0; i < 8; ++i) {
    int c = tid + i * 256;
    out[row * DDIM + c] = (hr[c] - mu) * inv * gamma[c] + beta[c];
  }
}

// ---------------------------------------------------------------------------
// launch
// ---------------------------------------------------------------------------
extern "C" void kernel_launch(void* const* d_in, const int* in_sizes, int n_in,
                              void* d_out, int out_size, void* d_ws, size_t ws_size,
                              hipStream_t stream) {
  (void)in_sizes; (void)n_in; (void)out_size; (void)ws_size;
  const float* x     = (const float*)d_in[0];
  const float* Q_w   = (const float*)d_in[1];
  const float* Q_b   = (const float*)d_in[2];
  const float* K_w   = (const float*)d_in[3];
  const float* K_b   = (const float*)d_in[4];
  const float* Lw    = (const float*)d_in[5];   // lo_proj_w [D,D]
  const float* Lb    = (const float*)d_in[6];
  const float* Lpw   = (const float*)d_in[7];   // lo_p_w [D,F]
  const float* Lpb   = (const float*)d_in[8];
  const float* Pw    = (const float*)d_in[9];   // proj_w [F,D]
  const float* Pb    = (const float*)d_in[10];
  const float* Ow    = (const float*)d_in[11];  // O_w [H,D]
  const float* Ob    = (const float*)d_in[12];
  const float* gamma = (const float*)d_in[13];
  const float* beta  = (const float*)d_in[14];
  float* out = (float*)d_out;

  // workspace carve-up (~346 MB)
  char* ws = (char*)d_ws;
  size_t off = 0;
  auto alloc = [&](size_t bytes) -> void* {
    void* p = ws + off;
    off += (bytes + 255) & ~(size_t)255;
    return p;
  };
  unsigned short* x_bf   = (unsigned short*)alloc((size_t)MTOT * DDIM * 2);
  unsigned short* QwT    = (unsigned short*)alloc((size_t)DDIM * FDIM * 2);  // [F][D]
  unsigned short* KwT    = (unsigned short*)alloc((size_t)DDIM * FDIM * 2);  // [F][D]
  unsigned short* LwT    = (unsigned short*)alloc((size_t)DDIM * DDIM * 2);  // [D][D]
  unsigned short* LpwT   = (unsigned short*)alloc((size_t)DDIM * FDIM * 2);  // [F][D]
  unsigned short* PwT    = (unsigned short*)alloc((size_t)FDIM * DDIM * 2);  // [D][F]
  unsigned short* OwT    = (unsigned short*)alloc((size_t)HDIM * DDIM * 2);  // [D][H]
  float*          sq     = (float*)alloc((size_t)MTOT * FDIM * 4);
  float*          score  = (float*)alloc((size_t)MTOT * FDIM * 4);  // -> ema in place
  float*          Vbuf   = (float*)alloc((size_t)MTOT * FDIM * 4);
  unsigned short* vh_bf  = (unsigned short*)alloc((size_t)MTOT * DDIM * 2);
  unsigned short* xc_bf  = (unsigned short*)alloc((size_t)MTOT * FDIM * 2);
  float*          outf   = (float*)alloc((size_t)MTOT * DDIM * 4);  // out, later h
  unsigned short* g_bf   = (unsigned short*)alloc((size_t)MTOT * HDIM * 2);

  dim3 blk(256);
  {  // activations: straight convert
    size_t n = (size_t)MTOT * DDIM;
    f32_to_bf16_kernel<<<dim3((unsigned)((n + 255) / 256)), blk, 0, stream>>>(x, x_bf, n);
  }
  auto cvtT = [&](const float* s, unsigned short* d, int K, int N) {  // [K][N]->[N][K]
    f32_to_bf16_T_kernel<<<dim3(N / 32, K / 32), blk, 0, stream>>>(s, d, K, N);
  };
  cvtT(Q_w, QwT, DDIM, FDIM);
  cvtT(K_w, KwT, DDIM, FDIM);
  cvtT(Lw,  LwT, DDIM, DDIM);
  cvtT(Lpw, LpwT, DDIM, FDIM);
  cvtT(Pw,  PwT, FDIM, DDIM);
  cvtT(Ow,  OwT, HDIM, DDIM);

  dim3 gF(FDIM / 128, MTOT / 64);  // N=128 GEMMs
  dim3 gD(DDIM / 128, MTOT / 64);  // N=2048 GEMMs

  // sq = sigmoid(x@Qw + Qb)
  gemm_bf16_wmma<EP_SIGMOID><<<gF, blk, 0, stream>>>(
      (const bf16_t*)x_bf, (const bf16_t*)QwT, Q_b, sq, nullptr, FDIM, DDIM);
  // score = sigmoid(x@Kw + Kb) * sq
  gemm_bf16_wmma<EP_SIGMOID_MUL><<<gF, blk, 0, stream>>>(
      (const bf16_t*)x_bf, (const bf16_t*)KwT, K_b, score, sq, FDIM, DDIM);
  // v_h = gelu(x@Lw + Lb) -> bf16
  gemm_bf16_wmma<EP_GELU_BF16><<<gD, blk, 0, stream>>>(
      (const bf16_t*)x_bf, (const bf16_t*)LwT, Lb, vh_bf, nullptr, DDIM, DDIM);
  // V = v_h@Lpw + Lpb
  gemm_bf16_wmma<EP_BIAS><<<gF, blk, 0, stream>>>(
      (const bf16_t*)vh_bf, (const bf16_t*)LpwT, Lpb, Vbuf, nullptr, FDIM, DDIM);

  ema_kernel<<<dim3(2), blk, 0, stream>>>(score);
  clipmul_kernel<<<dim3(MTOT / 8), blk, 0, stream>>>(score, Vbuf, xc_bf);

  // out = x_comb@Pw + Pb   (K = 128)
  gemm_bf16_wmma<EP_BIAS><<<gD, blk, 0, stream>>>(
      (const bf16_t*)xc_bf, (const bf16_t*)PwT, Pb, outf, nullptr, DDIM, FDIM);
  gate_kernel<<<dim3((unsigned)(((size_t)MTOT * HDIM) / 256)), blk, 0, stream>>>(outf, g_bf);
  // h = gated@Ow + Ob + residual(x)   (reuses outf; gate already consumed it)
  gemm_bf16_wmma<EP_BIAS_RESID><<<gD, blk, 0, stream>>>(
      (const bf16_t*)g_bf, (const bf16_t*)OwT, Ob, outf, x, DDIM, HDIM);

  ln_kernel<<<dim3(MTOT), blk, 0, stream>>>(outf, gamma, beta, out);
}
